// CausalSelfAttention_36945308680598
// MI455X (gfx1250) — matrix operand order repaired and checked by reading, not compiled, not run on hardware
//
#include <hip/hip_runtime.h>

// ---------------------------------------------------------------------------
// Types for WMMA fragments (CDNA5 wave32)
// ---------------------------------------------------------------------------
typedef __attribute__((ext_vector_type(16))) __bf16   bf16x16;
typedef __attribute__((ext_vector_type(8)))  float    f32x8;
typedef __attribute__((ext_vector_type(4)))  unsigned u32x4;
typedef unsigned short u16;

union BF16Frag { bf16x16 v; u32x4 q[2]; };

__device__ __forceinline__ f32x8 wmma_bf16(bf16x16 a, bf16x16 b, f32x8 c) {
    // D = A(16x32 bf16) * B(32x16 bf16) + C(16x16 f32)
    return __builtin_amdgcn_wmma_f32_16x16x32_bf16(
        /*neg_a=*/false, a, /*neg_b=*/false, b,
        /*c_mod=*/(short)0, c, /*reuse_a=*/false, /*reuse_b=*/false);
}

__device__ __forceinline__ u16 f2bf(float f) {
    unsigned u = __float_as_uint(f);
    unsigned r = (u + 0x7FFFu + ((u >> 16) & 1u)) >> 16;   // RNE
    return (u16)r;
}

// Problem constants
#define BATCH 4
#define T 2048
#define C 1024
#define H 16
#define D 64
#define NBUCKETS 32
#define MTOK (BATCH * T)            // 8192 flattened tokens

// ---------------------------------------------------------------------------
// Kernel 0: fp32 -> bf16 conversion
// ---------------------------------------------------------------------------
__global__ __launch_bounds__(256) void cvt_bf16_kernel(const float* __restrict__ in,
                                                       u16* __restrict__ out, int n) {
    int i = blockIdx.x * blockDim.x + threadIdx.x;
    if (i < n) out[i] = f2bf(in[i]);
}

// ---------------------------------------------------------------------------
// Shared GEMM core: one wave computes a 32x64 tile of  A[M,K] @ W[N,K]^T.
// Double-buffered fragment pipeline: loads for step k+32 are issued before
// the WMMAs of step k, so the scheduler can overlap VMEM with matrix ops.
// ---------------------------------------------------------------------------
struct Frags {
    BF16Frag a[2];   // two 16-row A fragments (rows r*16)
    BF16Frag b[4];   // four 16-col B fragments
};

__device__ __forceinline__ void load_frags(Frags& f, const u16* aptr0, const u16* aptr1,
                                           const u16* bbase, int k0) {
    f.a[0].q[0] = *(const u32x4*)(aptr0 + k0);
    f.a[0].q[1] = *(const u32x4*)(aptr0 + k0 + 16);
    f.a[1].q[0] = *(const u32x4*)(aptr1 + k0);
    f.a[1].q[1] = *(const u32x4*)(aptr1 + k0 + 16);
#pragma unroll
    for (int nt = 0; nt < 4; ++nt) {
        const u16* bp = bbase + (size_t)nt * 16 * C + k0;
        f.b[nt].q[0] = *(const u32x4*)(bp);
        f.b[nt].q[1] = *(const u32x4*)(bp + 16);
    }
}

__device__ __forceinline__ void gemm_wave_32x64(const u16* __restrict__ Ab,
                                                const u16* __restrict__ Wb,
                                                int row0, int col0, int m, int g,
                                                f32x8 (&c)[2][4]) {
    const u16* aptr0 = Ab + (size_t)(row0 + m) * C + g * 8;
    const u16* aptr1 = aptr0 + (size_t)16 * C;
    const u16* bbase = Wb + (size_t)(col0 + m) * C + g * 8;

    Frags buf[2];
    load_frags(buf[0], aptr0, aptr1, bbase, 0);
#pragma unroll 2
    for (int it = 0; it < C / 32; ++it) {
        const int cur = it & 1, nxt = cur ^ 1;
        const int k0 = it * 32;
        if (k0 + 32 < C) load_frags(buf[nxt], aptr0, aptr1, bbase, k0 + 32);
        // pull the A stream for upcoming iterations toward the WGP
        __builtin_prefetch(aptr0 + k0 + 512, 0, 1);
        __builtin_prefetch(aptr1 + k0 + 512, 0, 1);
#pragma unroll
        for (int nt = 0; nt < 4; ++nt) {
            c[0][nt] = wmma_bf16(buf[cur].a[0].v, buf[cur].b[nt].v, c[0][nt]);
            c[1][nt] = wmma_bf16(buf[cur].a[1].v, buf[cur].b[nt].v, c[1][nt]);
        }
    }
}

// ---------------------------------------------------------------------------
// Kernel 1: QKV GEMM  qkv = x @ W_attn^T + b_attn, scatter to Q/K/V layouts
//   grid = (MTOK/32 = 256, 3C/256 = 12), block = 128 (4 waves)
// ---------------------------------------------------------------------------
__global__ __launch_bounds__(128, 1)
void qkv_gemm_kernel(const u16* __restrict__ xb, const u16* __restrict__ Wab,
                     const float* __restrict__ b_attn,
                     u16* __restrict__ Qb, u16* __restrict__ Kb,
                     u16* __restrict__ Vtb) {
    const int lane = threadIdx.x & 31;
    const int wave = threadIdx.x >> 5;
    const int m = lane & 15, g = lane >> 4;
    const int row0 = blockIdx.x * 32;
    const int col0 = blockIdx.y * 256 + wave * 64;

    f32x8 c[2][4] = {};
    gemm_wave_32x64(xb, Wab, row0, col0, m, g, c);

#pragma unroll
    for (int nt = 0; nt < 4; ++nt) {
        const int col = col0 + nt * 16 + m;
        const float bias = b_attn[col];
        const int sect = col >> 10;          // 0=q 1=k 2=v
        const int cin = col & 1023;
        const int h = cin >> 6, d = cin & 63;
#pragma unroll
        for (int r = 0; r < 2; ++r)
#pragma unroll
            for (int i = 0; i < 8; ++i) {
                const int tok = row0 + r * 16 + i + 8 * g;
                const int b_ = tok >> 11;        // /T
                const int t = tok & (T - 1);
                const float v = c[r][nt][i] + bias;
                if (sect == 0)      Qb[(((size_t)(b_ * H + h) * T) + t) * D + d] = f2bf(v * 0.125f);
                else if (sect == 1) Kb[(((size_t)(b_ * H + h) * T) + t) * D + d] = f2bf(v);
                else                Vtb[(((size_t)(b_ * H + h) * D) + d) * T + t] = f2bf(v);
            }
    }
}

// ---------------------------------------------------------------------------
// Kernel 2: flash attention with T5 relative-position bias.
//   One wave handles a 32-query tile of one (b,h): K/V fragments per 32-key
//   step are loaded once and reused by both 16-row query halves.
//   grid = (T/32/4 = 16, B*H = 64), block = 128 (4 waves)
// ---------------------------------------------------------------------------
__device__ __forceinline__ int rp_bucket(int rp) {
    // unidirectional T5 bucket: rp = q - k >= 0
    if (rp < 16) return rp;
    int b = 16 + (int)(__logf((float)rp * (1.0f / 16.0f)) * (16.0f / 2.0794415416798357f));
    return b > 31 ? 31 : b;
}

__global__ __launch_bounds__(128, 1)
void attn_kernel(const u16* __restrict__ Qb, const u16* __restrict__ Kb,
                 const u16* __restrict__ Vtb, const float* __restrict__ rel_emb,
                 u16* __restrict__ yb) {
    __shared__ __align__(16) float s_rel[NBUCKETS * H];       // 2 KB
    __shared__ __align__(16) u16   s_p[4][2][16 * 32];        // 8 KB, per-wave P staging
    for (int i = threadIdx.x; i < NBUCKETS * H; i += blockDim.x) s_rel[i] = rel_emb[i];
    __syncthreads();

    const int lane = threadIdx.x & 31;
    const int wave = threadIdx.x >> 5;
    const int m = lane & 15, g = lane >> 4;
    const int qt = blockIdx.x * 4 + wave;      // 32-row query tile, 0..63
    const int bh = blockIdx.y;
    const int h = bh & 15;
    const int q0 = qt * 32;

    const u16* Qp = Qb + (size_t)bh * T * D;
    const u16* Kp = Kb + (size_t)bh * T * D;
    const u16* Vp = Vtb + (size_t)bh * D * T;

    // Q tile 32x64 as 2 (query-half) x 2 (k-chunk) A fragments; pre-scaled by 1/8
    BF16Frag qa[2][2];
#pragma unroll
    for (int r = 0; r < 2; ++r) {
        const u16* p = Qp + (size_t)(q0 + r * 16 + m) * D + g * 8;
        qa[r][0].q[0] = *(const u32x4*)(p);       qa[r][0].q[1] = *(const u32x4*)(p + 16);
        qa[r][1].q[0] = *(const u32x4*)(p + 32);  qa[r][1].q[1] = *(const u32x4*)(p + 48);
    }

    f32x8 o[2][4] = {};
    float mrow[2][8], lrow[2][8];
#pragma unroll
    for (int r = 0; r < 2; ++r)
#pragma unroll
        for (int i = 0; i < 8; ++i) { mrow[r][i] = -1e30f; lrow[r][i] = 0.f; }

    const int nsteps = qt + 1;                 // 32-key steps covering [0, q0+32)
    for (int kb = 0; kb < nsteps; ++kb) {
        const int kstep0 = kb * 32;
        if (kb + 1 < nsteps) {
            __builtin_prefetch(Kp + (size_t)(kstep0 + 32 + m) * D + g * 16, 0, 1);
            __builtin_prefetch(Vp + (size_t)m * T + kstep0 + 32 + g * 16, 0, 1);
        }
        // ---- K fragments for both 16-key subtiles (shared by both q halves) ----
        BF16Frag kf0[2], kf1[2];               // [js]: k-chunk 0..31 / 32..63 of D
#pragma unroll
        for (int js = 0; js < 2; ++js) {
            const u16* p = Kp + (size_t)(kstep0 + js * 16 + m) * D + g * 8;
            kf0[js].q[0] = *(const u32x4*)(p);        kf0[js].q[1] = *(const u32x4*)(p + 16);
            kf1[js].q[0] = *(const u32x4*)(p + 32);   kf1[js].q[1] = *(const u32x4*)(p + 48);
        }
        // ---- per query half: S = Q K^T, bias+mask, online softmax, stage P ----
#pragma unroll
        for (int r = 0; r < 2; ++r) {
            f32x8 s[2];
#pragma unroll
            for (int js = 0; js < 2; ++js) {
                f32x8 acc = {};
                acc = wmma_bf16(qa[r][0].v, kf0[js].v, acc);
                acc = wmma_bf16(qa[r][1].v, kf1[js].v, acc);
                s[js] = acc;
            }
            u16* myp = s_p[wave][r];
            float alpha[8];
#pragma unroll
            for (int i = 0; i < 8; ++i) {
                const int q = q0 + r * 16 + i + 8 * g;
                const int k0i = kstep0 + m;
                const int k1i = kstep0 + 16 + m;
                float s0 = s[0][i], s1 = s[1][i];
                if (k0i > q) s0 = -1e30f;
                else         s0 += s_rel[rp_bucket(q - k0i) * H + h];
                if (k1i > q) s1 = -1e30f;
                else         s1 += s_rel[rp_bucket(q - k1i) * H + h];
                float mx = fmaxf(s0, s1);
#pragma unroll
                for (int d_ = 8; d_ >= 1; d_ >>= 1) mx = fmaxf(mx, __shfl_xor(mx, d_, 32));
                const float mnew = fmaxf(mrow[r][i], mx);
                const float a_ = __expf(mrow[r][i] - mnew);
                const float p0 = __expf(s0 - mnew);
                const float p1 = __expf(s1 - mnew);
                float rs = p0 + p1;
#pragma unroll
                for (int d_ = 8; d_ >= 1; d_ >>= 1) rs += __shfl_xor(rs, d_, 32);
                lrow[r][i] = lrow[r][i] * a_ + rs;
                mrow[r][i] = mnew;
                alpha[i] = a_;
                const int rr = i + 8 * g;
                myp[rr * 32 + m]      = f2bf(p0);   // C-layout -> LDS (row-major 16x32)
                myp[rr * 32 + 16 + m] = f2bf(p1);
            }
#pragma unroll
            for (int nt = 0; nt < 4; ++nt)
#pragma unroll
                for (int i = 0; i < 8; ++i) o[r][nt][i] *= alpha[i];
        }
        // ---- P tiles back as A fragments (in-order wave-local LDS) ----
        BF16Frag pa[2];
#pragma unroll
        for (int r = 0; r < 2; ++r) {
            const u16* p = s_p[wave][r] + m * 32 + g * 8;
            pa[r].q[0] = *(const u32x4*)(p);
            pa[r].q[1] = *(const u32x4*)(p + 16);
        }
        // ---- O += P * V : V fragment loaded once, used by both q halves ----
#pragma unroll
        for (int nt = 0; nt < 4; ++nt) {
            const u16* p = Vp + (size_t)(nt * 16 + m) * T + kstep0 + g * 8;
            BF16Frag vb;
            vb.q[0] = *(const u32x4*)(p);
            vb.q[1] = *(const u32x4*)(p + 16);
            o[0][nt] = wmma_bf16(pa[0].v, vb.v, o[0][nt]);
            o[1][nt] = wmma_bf16(pa[1].v, vb.v, o[1][nt]);
        }
    }
    // ---- finalize: divide by softmax denominator, store bf16 [B,T,C] ----
    const int b_ = bh >> 4;
#pragma unroll
    for (int r = 0; r < 2; ++r) {
        float inv[8];
#pragma unroll
        for (int i = 0; i < 8; ++i) inv[i] = 1.0f / lrow[r][i];
#pragma unroll
        for (int nt = 0; nt < 4; ++nt) {
            const int col = h * D + nt * 16 + m;
#pragma unroll
            for (int i = 0; i < 8; ++i) {
                const int tok = b_ * T + q0 + r * 16 + i + 8 * g;
                yb[(size_t)tok * C + col] = f2bf(o[r][nt][i] * inv[i]);
            }
        }
    }
}

// ---------------------------------------------------------------------------
// Kernel 3: projection GEMM  y = y_att @ W_proj^T + b_proj  (fp32 out)
//   grid = (MTOK/32 = 256, C/256 = 4), block = 128 (4 waves)
// ---------------------------------------------------------------------------
__global__ __launch_bounds__(128, 1)
void proj_gemm_kernel(const u16* __restrict__ yb, const u16* __restrict__ Wpb,
                      const float* __restrict__ b_proj, float* __restrict__ out) {
    const int lane = threadIdx.x & 31;
    const int wave = threadIdx.x >> 5;
    const int m = lane & 15, g = lane >> 4;
    const int row0 = blockIdx.x * 32;
    const int col0 = blockIdx.y * 256 + wave * 64;

    f32x8 c[2][4] = {};
    gemm_wave_32x64(yb, Wpb, row0, col0, m, g, c);

#pragma unroll
    for (int nt = 0; nt < 4; ++nt) {
        const int col = col0 + nt * 16 + m;
        const float bias = b_proj[col];
#pragma unroll
        for (int r = 0; r < 2; ++r)
#pragma unroll
            for (int i = 0; i < 8; ++i) {
                const int row = row0 + r * 16 + i + 8 * g;
                out[(size_t)row * C + col] = c[r][nt][i] + bias;
            }
    }
}

// ---------------------------------------------------------------------------
// Host launcher
// ---------------------------------------------------------------------------
extern "C" void kernel_launch(void* const* d_in, const int* in_sizes, int n_in,
                              void* d_out, int out_size, void* d_ws, size_t ws_size,
                              hipStream_t stream) {
    const float* x      = (const float*)d_in[0];
    const float* W_attn = (const float*)d_in[1];
    const float* b_attn = (const float*)d_in[2];
    const float* W_proj = (const float*)d_in[3];
    const float* b_proj = (const float*)d_in[4];
    const float* rel    = (const float*)d_in[5];

    char* ws = (char*)d_ws;
    size_t off = 0;
    auto carve = [&](size_t bytes) -> void* {
        void* p = ws + off;
        off += (bytes + 255) & ~(size_t)255;
        return p;
    };
    u16* xb  = (u16*)carve((size_t)MTOK * C * 2);          // 16 MB
    u16* Wab = (u16*)carve((size_t)3 * C * C * 2);         //  6 MB
    u16* Wpb = (u16*)carve((size_t)C * C * 2);             //  2 MB
    u16* Qb  = (u16*)carve((size_t)BATCH * H * T * D * 2); // 16 MB
    u16* Kb  = (u16*)carve((size_t)BATCH * H * T * D * 2); // 16 MB
    u16* Vtb = (u16*)carve((size_t)BATCH * H * D * T * 2); // 16 MB
    u16* yb  = (u16*)carve((size_t)MTOK * C * 2);          // 16 MB

    const int nx = MTOK * C, nwa = 3 * C * C, nwp = C * C;
    cvt_bf16_kernel<<<(nx + 255) / 256, 256, 0, stream>>>(x, xb, nx);
    cvt_bf16_kernel<<<(nwa + 255) / 256, 256, 0, stream>>>(W_attn, Wab, nwa);
    cvt_bf16_kernel<<<(nwp + 255) / 256, 256, 0, stream>>>(W_proj, Wpb, nwp);

    qkv_gemm_kernel<<<dim3(MTOK / 32, (3 * C) / 256), 128, 0, stream>>>(
        xb, Wab, b_attn, Qb, Kb, Vtb);

    attn_kernel<<<dim3((T / 32) / 4, BATCH * H), 128, 0, stream>>>(
        Qb, Kb, Vtb, rel, yb);

    proj_gemm_kernel<<<dim3(MTOK / 32, C / 256), 128, 0, stream>>>(
        yb, Wpb, b_proj, (float*)d_out);
}
